// hist_loss_11029476016659
// MI455X (gfx1250) — compile-verified, matrix operand order checked by reading, not run on hardware
//
#include <hip/hip_runtime.h>
#include <hip/hip_bf16.h>

// ---------------------------------------------------------------------------
// KDE-histogram exposure loss for MI455X (gfx1250, wave32).
//
// Pipeline:
//   kernel 1 (kde_hist_kernel): 4 samples x 26 regions (25 crops + border
//     remainder) x 4 row-chunks. Each block computes gray = gamma(exposure)
//     inline (v_log_f32 / v_exp_f32), then a *windowed* KDE histogram
//     (16 bins around round(g*127); farther bins contribute < 1e-30) into
//     per-wave LDS histograms via ds_add_f32, then writes f32 partials to
//     d_ws (disjoint slots -> deterministic).
//   kernel 2 (hist_loss_kernel): one wave per sample. Merges the 26 region
//     histograms into the full-image histogram with v_wmma_f32_16x16x32_f16
//     (A = ones 16x32, B rows = region histograms, D rows = column sums),
//     then computes 1 - cos(pdf, uniform) for the global histogram and the
//     25 crop histograms, and averages.
// ---------------------------------------------------------------------------

#define NBINS  128
#define NREG   26     // 25 crops + 1 border remainder region
#define NCHUNK 4      // row-chunks per region (more blocks in flight)

typedef __attribute__((ext_vector_type(16))) _Float16 v16h;
typedef __attribute__((ext_vector_type(8)))  float    v8f;

// Raw hardware transcendentals (base-2): v_exp_f32 / v_log_f32.
__device__ __forceinline__ float fast_exp2(float x) {
    return __builtin_amdgcn_exp2f(x);
}
__device__ __forceinline__ float fast_log2(float x) {
    return __builtin_amdgcn_logf(x);
}
__device__ __forceinline__ float fast_exp(float x) {
    return __builtin_amdgcn_exp2f(x * 1.4426950408889634f); // exp(x)=2^(x*log2 e)
}

__device__ __forceinline__ float gamma_gray(float r, float g, float b,
                                            float e0, float e1, float e2) {
    const float INVG = 0.45454545454545453f;  // 1/2.2
    float v0 = fminf(fmaxf(r * e0, 1e-12f), 4095.0f) * (1.0f / 4095.0f);
    float v1 = fminf(fmaxf(g * e1, 1e-12f), 4095.0f) * (1.0f / 4095.0f);
    float v2 = fminf(fmaxf(b * e2, 1e-12f), 4095.0f) * (1.0f / 4095.0f);
    float l0 = fminf(fast_exp2(fast_log2(v0) * INVG), 1.0f);
    float l1 = fminf(fast_exp2(fast_log2(v1) * INVG), 1.0f);
    float l2 = fminf(fast_exp2(fast_log2(v2) * INVG), 1.0f);
    return 0.299f * l0 + 0.587f * l1 + 0.114f * l2;
}

__global__ __launch_bounds__(256) void kde_hist_kernel(
        const float* __restrict__ hdr, const float* __restrict__ param,
        float* __restrict__ partial) {
    const int blk   = blockIdx.x;            // 0 .. 4*26*4 - 1
    const int chunk = blk & (NCHUNK - 1);
    const int sr    = blk >> 2;              // sample*26 + region
    const int s     = sr / NREG;
    const int r     = sr - s * NREG;
    const int tid   = threadIdx.x;
    const int wid   = tid >> 5;              // wave id (wave32 -> 8 waves)

    __shared__ float whist[8][NBINS];        // per-wave private histograms
    for (int i = tid; i < 8 * NBINS; i += 256) ((float*)whist)[i] = 0.0f;
    __syncthreads();

    const float e0 = fast_exp2(param[s * 3 + 0]);
    const float e1 = fast_exp2(param[s * 3 + 1]);
    const float e2 = fast_exp2(param[s * 3 + 2]);
    const float* base = hdr + (size_t)s * (3 * 512 * 512);

    const int npix = (r < 25) ? (102 * 102) : 2044;       // border = 2044 px
    const int seg  = (npix + NCHUNK - 1) / NCHUNK;
    const int pbeg = chunk * seg;
    const int pend = (npix < pbeg + seg) ? npix : (pbeg + seg);
    const int cy = r / 5, cx = r - cy * 5;                // valid for r < 25

    for (int p = pbeg + tid; p < pend; p += 256) {
        int y, x;
        if (r < 25) {
            int py = p / 102;
            int px = p - py * 102;
            y = cy * 102 + py;
            x = cx * 102 + px;
        } else {                     // remainder: rows 510..511 + cols 510..511
            if (p < 1024) { y = 510 + (p >> 9); x = p & 511; }
            else { int q = p - 1024; y = q >> 1; x = 510 + (q & 1); }
        }
        const int idx = y * 512 + x;
        __builtin_prefetch(base + idx + 2048, 0, 0);      // global_prefetch_b8
        const float g = gamma_gray(base[idx],
                                   base[idx + 262144],
                                   base[idx + 524288], e0, e1, e2);
        // 16-bin window around nearest bin; outside the window the Gaussian
        // is <= exp(-69.7) and cannot perturb an f32 sum.
        const int j0 = (int)floorf(fmaf(g, 127.0f, 0.5f)) - 8;
#pragma unroll
        for (int i = 0; i < 16; ++i) {
            const int j = j0 + i;
            if (j >= 0 && j < NBINS) {
                const float d = fmaf((float)j, -(1.0f / 127.0f), g);
                atomicAdd(&whist[wid][j], fast_exp(-20000.0f * d * d)); // ds_add_f32
            }
        }
    }
    __syncthreads();
    for (int j = tid; j < NBINS; j += 256) {
        float acc = 0.0f;
#pragma unroll
        for (int w = 0; w < 8; ++w) acc += whist[w][j];
        partial[(size_t)(sr * NCHUNK + chunk) * NBINS + j] = acc;
    }
}

__global__ __launch_bounds__(128) void hist_loss_kernel(
        const float* __restrict__ partial, float* __restrict__ out) {
    const int tid  = threadIdx.x;
    const int s    = tid >> 5;               // one wave per sample
    const int lane = tid & 31;
    __shared__ float losses[4];

    const int c    = lane & 15;              // B-matrix column = bin-in-tile
    const int kgrp = (lane >> 4) << 4;       // lanes 16..31 carry K rows 16..31

    v16h ones;
#pragma unroll
    for (int i = 0; i < 16; ++i) ones[i] = (_Float16)1.0f;

    const float S = 0.015625f, invS = 64.0f; // pre-scale for f16 headroom
    const float m = (lane < 16) ? 1.0f : 0.0f;  // N columns duplicated; count once
    float sumG = 0.0f, sqG = 0.0f;

    // Global histogram = ones(16x32) x B(32x16), 8 column tiles of 16 bins.
    // B rows 0..25 = region histograms (chunk-merged), rows 26..31 = 0.
#pragma unroll
    for (int t = 0; t < 8; ++t) {
        v16h bm;
#pragma unroll
        for (int i = 0; i < 16; ++i) {
            const int k = kgrp + i;
            float v = 0.0f;
            if (k < NREG) {
                const float* hp = partial
                    + (size_t)((s * NREG + k) * NCHUNK) * NBINS + t * 16 + c;
                v = (hp[0] + hp[NBINS] + hp[2 * NBINS] + hp[3 * NBINS]) * S;
            }
            bm[i] = (_Float16)v;
        }
        v8f cz = {};
        v8f dmat = __builtin_amdgcn_wmma_f32_16x16x32_f16(
            false, ones, false, bm, (short)0, cz, false, false);
        const float gbin = dmat[0] * invS;   // col-sum for bin t*16 + c
        sumG += m * gbin;
        sqG  += m * gbin * gbin;
    }
#pragma unroll
    for (int off = 16; off > 0; off >>= 1) {
        sumG += __shfl_xor(sumG, off, 32);
        sqG  += __shfl_xor(sqG,  off, 32);
    }

    // cosine vs uniform: a = h/(H + N*1e-10); b_j = 1/128, ||b|| = 1/sqrt(128)
    const float nb = 0.08838834764831845f;   // 1/sqrt(128)
    const float T1 = sumG + 262144.0f * 1e-10f;
    const float cos1 = ((sumG / T1) * (1.0f / 128.0f))
                     / (fmaxf(sqrtf(sqG) / T1, 1e-8f) * nb);
    float loss = 1.0f - cos1;

    float l5 = 0.0f;
    for (int r = 0; r < 25; ++r) {
        float s0 = 0.0f, q0 = 0.0f;
#pragma unroll
        for (int jj = 0; jj < 4; ++jj) {
            const float* hp = partial
                + (size_t)((s * NREG + r) * NCHUNK) * NBINS + jj * 32 + lane;
            const float v = hp[0] + hp[NBINS] + hp[2 * NBINS] + hp[3 * NBINS];
            s0 += v; q0 += v * v;
        }
#pragma unroll
        for (int off = 16; off > 0; off >>= 1) {
            s0 += __shfl_xor(s0, off, 32);
            q0 += __shfl_xor(q0, off, 32);
        }
        const float T  = s0 + 10404.0f * 1e-10f;
        const float cs = ((s0 / T) * (1.0f / 128.0f))
                       / (fmaxf(sqrtf(q0) / T, 1e-8f) * nb);
        l5 += 1.0f - cs;
    }
    loss += l5 * 0.04f;                      // mean over 25 crops

    if (lane == 0) losses[s] = loss;
    __syncthreads();
    if (tid == 0)
        out[0] = 0.25f * (losses[0] + losses[1] + losses[2] + losses[3]);
}

extern "C" void kernel_launch(void* const* d_in, const int* in_sizes, int n_in,
                              void* d_out, int out_size, void* d_ws, size_t ws_size,
                              hipStream_t stream) {
    (void)in_sizes; (void)n_in; (void)out_size; (void)ws_size;
    const float* hdr   = (const float*)d_in[0];   // [4,3,512,512] f32
    const float* param = (const float*)d_in[1];   // [4,3] f32
    float* partial = (float*)d_ws;                // 4*26*4*128 f32 = 208 KB
    float* out     = (float*)d_out;               // 1 f32

    kde_hist_kernel<<<4 * NREG * NCHUNK, 256, 0, stream>>>(hdr, param, partial);
    hist_loss_kernel<<<1, 128, 0, stream>>>(partial, out);
}
